// SphericalFourierNeuralOperatorModel_11175504904529
// MI455X (gfx1250) — matrix operand order repaired and checked by reading, not compiled, not run on hardware
//
#include <hip/hip_runtime.h>
#include <cmath>

// ---------------- problem constants ----------------
#define BB   8
#define INC  4
#define HH   180
#define WW   360
#define CC   64
#define DEPTH 6
#define ML   32
#define MW   64
#define NH   360
#define SPIX (BB*HH*WW)           // 518400 pixels
#define SC   ((long long)SPIX*CC) // 33,177,600 activation elems

typedef __bf16 bf16;
typedef __attribute__((ext_vector_type(16))) __bf16 v16bf;
typedef __attribute__((ext_vector_type(8)))  __bf16 v8bf;
typedef __attribute__((ext_vector_type(4)))  __bf16 v4bf;
typedef __attribute__((ext_vector_type(8)))  float  v8f;

__device__ __forceinline__ float gelu_exact(float x) {
    return 0.5f * x * (1.0f + erff(x * 0.70710678118654752f));
}

// =====================================================================
// Generic batched GEMM, bf16 inputs, f32 accumulate via WMMA.
// C[b] = A[b] @ B[b]   (A: MxK row-major, B: KxN row-major)
// mode: 0 = store f32, 1 = store bf16, 2 = accumulate into f32 C
// grid: (ceil(M/64), ceil(N/64), batch), block: 128 threads (4 waves)
//
// Interior tiles: A staged with gfx1250 async global->LDS DMA
// (global_load_async_to_lds_b64, ASYNCcnt), B staged with coalesced b64
// vector loads + register 4x4 transpose. Edge tiles: clamped-address
// branch-free scalar loads.
// =====================================================================
__global__ __launch_bounds__(128)
void gemm_bf16_wmma(const bf16* __restrict__ A, long long sA, int lda,
                    const bf16* __restrict__ B, long long sB, int ldb,
                    void* __restrict__ Cv, long long sC, int ldc,
                    int M, int N, int K, int mode)
{
    const int batch = blockIdx.z;
    const int row0  = blockIdx.x * 64;
    const int col0  = blockIdx.y * 64;
    A += (long long)batch * sA;
    B += (long long)batch * sB;

    __shared__ __align__(16) bf16 As[64][32];   // M x K tile
    __shared__ __align__(16) bf16 Bs[64][32];   // transposed: [n][k]

    const int tid  = threadIdx.x;
    const int wv   = tid >> 5;
    const int lane = tid & 31;
    const int l    = lane & 15;
    const int hi   = lane >> 4;
    const int m0   = wv * 16;

    const unsigned ldsA0 = (unsigned)(uintptr_t)(void*)&As[0][0];

    v8f acc[4];
    for (int s = 0; s < 4; ++s)
        for (int e = 0; e < 8; ++e) acc[s][e] = 0.0f;

    const bool fullM = (row0 + 64 <= M);
    const bool fullN = (col0 + 64 <= N);

    const int nk = (K + 31) >> 5;
    for (int kt = 0; kt < nk; ++kt) {
        const int k0 = kt << 5;
        const bool fullK = (k0 + 32 <= K);

        if (kt + 1 < nk) {  // speculative prefetch of next tiles -> global_prefetch_b8
            __builtin_prefetch(&A[(long long)(row0 + (tid >> 3)) * lda + k0 + 32], 0, 0);
            __builtin_prefetch(&B[(long long)(k0 + 32 + (tid >> 5)) * ldb + col0], 0, 0);
        }

        // ---------------- stage A tile (64x32) ----------------
        if (fullM && fullK) {
            // async DMA straight into LDS: 512 x 8B chunks, 4 per thread
            #pragma unroll
            for (int j = 0; j < 4; ++j) {
                int q  = tid + 128 * j;             // chunk id
                int mm = q >> 3, kc = (q & 7) * 4;  // 8 chunks per row
                const bf16* gp = A + (long long)(row0 + mm) * lda + (k0 + kc);
                unsigned la = ldsA0 + (unsigned)(mm * 64 + kc * 2);
                asm volatile("global_load_async_to_lds_b64 %0, %1, off"
                             :: "v"(la), "v"(gp) : "memory");
            }
        } else {
            // branch-free clamped loads
            #pragma unroll
            for (int j = 0; j < 16; ++j) {
                int i  = tid + j * 128;
                int mm = i >> 5, kk = i & 31;
                int gr = row0 + mm, gk = k0 + kk;
                int cr = gr < M ? gr : M - 1;
                int ck = gk < K ? gk : K - 1;
                bf16 v = A[(long long)cr * lda + ck];
                As[mm][kk] = (gr < M && gk < K) ? v : (bf16)0.0f;
            }
        }

        // ---------------- stage B tile transposed (Bs[n][k]) ----------------
        if (fullN && fullK) {
            // each thread: one 4x4 block, coalesced b64 loads + register transpose
            int bn = tid & 15, bk = tid >> 4;
            int n0b = bn * 4, k0b = bk * 4;
            v4bf r0 = *(const v4bf*)(B + (long long)(k0 + k0b + 0) * ldb + (col0 + n0b));
            v4bf r1 = *(const v4bf*)(B + (long long)(k0 + k0b + 1) * ldb + (col0 + n0b));
            v4bf r2 = *(const v4bf*)(B + (long long)(k0 + k0b + 2) * ldb + (col0 + n0b));
            v4bf r3 = *(const v4bf*)(B + (long long)(k0 + k0b + 3) * ldb + (col0 + n0b));
            #pragma unroll
            for (int j = 0; j < 4; ++j) {
                v4bf t = { r0[j], r1[j], r2[j], r3[j] };
                *(v4bf*)&Bs[n0b + j][k0b] = t;
            }
        } else {
            #pragma unroll
            for (int j = 0; j < 16; ++j) {
                int i  = tid + j * 128;
                int nn = i & 63, kk = i >> 6;
                int gk = k0 + kk, gc = col0 + nn;
                int ck = gk < K ? gk : K - 1;
                int cc = gc < N ? gc : N - 1;
                bf16 v = B[(long long)ck * ldb + cc];
                Bs[nn][kk] = (gk < K && gc < N) ? v : (bf16)0.0f;
            }
        }

        if (fullM && fullK)
            asm volatile("s_wait_asynccnt 0" ::: "memory");
        __syncthreads();

        // A fragment: 16-bit A 16x32 layout (ISA 7.12.2):
        // lane m = l; elems 0..7 -> k = hi*8+e ; elems 8..15 -> k = 16+hi*8+e
        v16bf afr;
        {
            v8bf a0 = *(const v8bf*)&As[m0 + l][hi * 8];
            v8bf a1 = *(const v8bf*)&As[m0 + l][16 + hi * 8];
            #pragma unroll
            for (int e = 0; e < 8; ++e) { afr[e] = a0[e]; afr[8 + e] = a1[e]; }
        }
        // 4 N-subtiles of 16
        #pragma unroll
        for (int s = 0; s < 4; ++s) {
            // B fragment: lane n = l; elems 0..7 -> k = hi*8+e ; elems 8..15 -> k = 16+hi*8+e
            v16bf bfr;
            v8bf b0 = *(const v8bf*)&Bs[s * 16 + l][hi * 8];
            v8bf b1 = *(const v8bf*)&Bs[s * 16 + l][16 + hi * 8];
            #pragma unroll
            for (int e = 0; e < 8; ++e) { bfr[e] = b0[e]; bfr[8 + e] = b1[e]; }

            acc[s] = __builtin_amdgcn_wmma_f32_16x16x32_bf16(
                         false, afr, false, bfr, (short)0, acc[s], false, false);
        }
        __syncthreads();
    }

    // ---- store: C/D layout: VGPR v -> row m0+hi*8+v, col = n-subtile + l ----
    const long long cb = (long long)batch * sC;
    if (fullM && fullN) {
        #pragma unroll
        for (int s = 0; s < 4; ++s) {
            int col = col0 + s * 16 + l;
            #pragma unroll
            for (int v = 0; v < 8; ++v) {
                int row = row0 + m0 + hi * 8 + v;
                long long idx = cb + (long long)row * ldc + col;
                if (mode == 1)      ((bf16*)Cv)[idx]  = (bf16)acc[s][v];
                else if (mode == 2) ((float*)Cv)[idx] += acc[s][v];
                else                ((float*)Cv)[idx] = acc[s][v];
            }
        }
    } else {
        for (int s = 0; s < 4; ++s) {
            int col = col0 + s * 16 + l;
            if (col >= N) continue;
            for (int v = 0; v < 8; ++v) {
                int row = row0 + m0 + hi * 8 + v;
                if (row >= M) continue;
                long long idx = cb + (long long)row * ldc + col;
                if (mode == 1)      ((bf16*)Cv)[idx]  = (bf16)acc[s][v];
                else if (mode == 2) ((float*)Cv)[idx] += acc[s][v];
                else                ((float*)Cv)[idx] = acc[s][v];
            }
        }
    }
}

// =====================================================================
// Basis build: truncated-DFT matrices (ortho norm) as bf16.
// =====================================================================
__global__ void build_basis(bf16* Fw, bf16* Fh, bf16* GhI, bf16* Gw)
{
    const float PI2 = 6.28318530717958647692f;
    int t = blockIdx.x * 256 + threadIdx.x;
    if (t < 128 * 360) {   // Fw: rows 0..63 cos, 64..127 -sin, /sqrt(360)
        int r = t / 360, w = t % 360;
        int m = r & 63;
        float th = PI2 * (float)((m * w) % 360) / 360.0f;
        float sc = 1.0f / sqrtf(360.0f);
        Fw[t] = (bf16)(((r >= 64) ? -sinf(th) : cosf(th)) * sc);
        return;
    }
    t -= 128 * 360;
    if (t < 64 * 180) {    // Fh: rows 0..31 cos, 32..63 -sin, /sqrt(180)
        int r = t / 180, h = t % 180;
        int m = r & 31;
        float th = PI2 * (float)((m * h) % 180) / 180.0f;
        float sc = 1.0f / sqrtf(180.0f);
        Fh[t] = (bf16)(((r >= 32) ? -sinf(th) : cosf(th)) * sc);
        return;
    }
    t -= 64 * 180;
    if (t < 360 * 64) {    // GhI: interleaved re/im rows of inverse-H DFT
        int row = t / 64, j = t % 64;
        int h = row >> 1, p = row & 1;
        int ml = j & 31, part = j >> 5;
        float th = PI2 * (float)((ml * h) % 180) / 180.0f;
        float sc = 1.0f / sqrtf(180.0f);
        float v;
        if (p == 0) v = (part == 0) ?  cosf(th) : -sinf(th);
        else        v = (part == 0) ?  sinf(th) :  cosf(th);
        GhI[t] = (bf16)(v * sc);
        return;
    }
    t -= 360 * 64;
    if (t < 360 * 128) {   // Gw: inverse-W real DFT, alpha_m doubling
        int w = t / 128, j = t % 128;
        int m = j & 63, part = j >> 6;
        float th = PI2 * (float)((m * w) % 360) / 360.0f;
        float a  = (m == 0) ? 1.0f : 2.0f;
        float sc = a / sqrtf(360.0f);
        Gw[t] = (bf16)(((part == 0) ? cosf(th) : -sinf(th)) * sc);
        return;
    }
}

// =====================================================================
// Input projection: x (B,INC,H,W) -> activation (pixel-major, C-last)
// =====================================================================
__global__ void input_proj(const float* __restrict__ x,
                           const float* __restrict__ in_w,
                           const float* __restrict__ in_b,
                           float* __restrict__ xact, bf16* __restrict__ xb)
{
    long long t = (long long)blockIdx.x * 256 + threadIdx.x;
    if (t >= SC) return;
    int c = (int)(t & 63);
    long long s = t >> 6;
    int b  = (int)(s / (HH * WW));
    int hw = (int)(s % (HH * WW));
    const float* xp = x + ((long long)b * INC) * (HH * WW) + hw;
    float v = in_b[c];
    #pragma unroll
    for (int i = 0; i < INC; ++i) v += xp[(long long)i * (HH * WW)] * in_w[c * INC + i];
    xact[t] = v;
    xb[t]   = (bf16)v;
}

// =====================================================================
// Per-depth packing kernels
// =====================================================================
__global__ void pack_linw(const float* __restrict__ lin_w_d, bf16* __restrict__ linT)
{
    int t = blockIdx.x * 256 + threadIdx.x;
    if (t >= CC * CC) return;
    int k = t >> 6, n = t & 63;               // linT[k][n] = lin_w[n][k]
    linT[t] = (bf16)lin_w_d[n * CC + k];
}

// Wpack[mode][128][128] = [[Wre, Wim], [-Wim, Wre]] per mode (ml,mw)
__global__ void pack_wmix(const float* __restrict__ fw_d, bf16* __restrict__ Wpack)
{
    long long t = (long long)blockIdx.x * 256 + threadIdx.x;
    if (t >= (long long)2048 * 128 * 128) return;
    int nn = (int)(t & 127);
    int kk = (int)((t >> 7) & 127);
    int mode = (int)(t >> 14);
    int ml = mode >> 6, mw = mode & 63;
    int c = kk & 63, o = nn & 63;
    int comp; float sgn = 1.0f;
    if (kk < 64)  comp = (nn < 64) ? 0 : 1;
    else { if (nn < 64) { comp = 1; sgn = -1.0f; } else comp = 0; }
    long long fi = ((((long long)c * CC + o) * ML + ml) * MW + mw) * 2 + comp;
    Wpack[t] = (bf16)(sgn * fw_d[fi]);
}

// combine F2 output (f32) into Zpack bf16: A-operand of MIX
__global__ void combine_Z(const float* __restrict__ C2, bf16* __restrict__ Zpack)
{
    long long t = (long long)blockIdx.x * 256 + threadIdx.x;
    if (t >= (long long)2048 * 8 * 128) return;
    int k = (int)(t & 127);
    int b = (int)((t >> 7) & 7);
    int mode = (int)(t >> 10);
    int ml = mode >> 6, mw = mode & 63;
    int c = k & 63;
    long long base = (long long)b * 64 * 8192;
    long long colre = (long long)mw * 64 + c;
    long long colim = 4096 + colre;
    float v;
    if (k < 64)
        v = C2[base + (long long)ml * 8192 + colre] - C2[base + (long long)(32 + ml) * 8192 + colim];
    else
        v = C2[base + (long long)ml * 8192 + colim] + C2[base + (long long)(32 + ml) * 8192 + colre];
    Zpack[t] = (bf16)v;
}

// Bmat for inverse-H GEMM: Bmat[b][k(64)][mw*64+o] from Mout[mode][b][128]
__global__ void pack_I1(const bf16* __restrict__ Mout, bf16* __restrict__ Bmat)
{
    long long t = (long long)blockIdx.x * 256 + threadIdx.x;
    if (t >= (long long)8 * 64 * 4096) return;
    int n = (int)(t & 4095);
    int k = (int)((t >> 12) & 63);
    int b = (int)(t >> 18);
    int mw = n >> 6, o = n & 63;
    bf16 v;
    if (k < 32) v = Mout[(((long long)(k * 64 + mw)) * 8 + b) * 128 + o];
    else        v = Mout[(((long long)((k - 32) * 64 + mw)) * 8 + b) * 128 + 64 + o];
    Bmat[t] = v;
}

// =====================================================================
// Pointwise: x = gelu(LN(gelu(x1 + bias))) ; one wave per pixel
// =====================================================================
__global__ __launch_bounds__(256)
void pw_block(const float* __restrict__ x1,
              const float* __restrict__ lin_b_d,
              const float* __restrict__ ln_g_d,
              const float* __restrict__ ln_b_d,
              float* __restrict__ xact, bf16* __restrict__ xb)
{
    int pix  = blockIdx.x * 8 + (threadIdx.x >> 5);
    int lane = threadIdx.x & 31;
    const float* p = x1 + (long long)pix * CC;
    float v0 = gelu_exact(p[lane]      + lin_b_d[lane]);
    float v1 = gelu_exact(p[lane + 32] + lin_b_d[lane + 32]);
    float s = v0 + v1, ss = v0 * v0 + v1 * v1;
    #pragma unroll
    for (int o = 16; o; o >>= 1) { s += __shfl_xor(s, o, 32); ss += __shfl_xor(ss, o, 32); }
    float mu  = s * (1.0f / 64.0f);
    float var = ss * (1.0f / 64.0f) - mu * mu;
    float ri  = rsqrtf(var + 1e-5f);
    float o0 = gelu_exact((v0 - mu) * ri * ln_g_d[lane]      + ln_b_d[lane]);
    float o1 = gelu_exact((v1 - mu) * ri * ln_g_d[lane + 32] + ln_b_d[lane + 32]);
    long long t0 = (long long)pix * CC + lane;
    xact[t0] = o0;      xact[t0 + 32] = o1;
    xb[t0]   = (bf16)o0; xb[t0 + 32]  = (bf16)o1;
}

// =====================================================================
// Head: per-pixel gathered (OC x C) matvec; one wave per pixel
// =====================================================================
__global__ __launch_bounds__(256)
void head_block(const float* __restrict__ xact, const int* __restrict__ grid,
                const float* __restrict__ head_w, const float* __restrict__ head_b,
                float* __restrict__ out)
{
    int pix  = blockIdx.x * 8 + (threadIdx.x >> 5);
    int lane = threadIdx.x & 31;
    int idx = grid[pix];
    idx = idx < 0 ? 0 : (idx > NH - 1 ? NH - 1 : idx);
    const float* xp = xact + (long long)pix * CC;
    float x0 = xp[lane], x1 = xp[lane + 32];
    const float* w0 = head_w + ((long long)idx * 2 + 0) * CC;
    const float* w1 = head_w + ((long long)idx * 2 + 1) * CC;
    float a = x0 * w0[lane] + x1 * w0[lane + 32];
    float b = x0 * w1[lane] + x1 * w1[lane + 32];
    #pragma unroll
    for (int o = 16; o; o >>= 1) { a += __shfl_xor(a, o, 32); b += __shfl_xor(b, o, 32); }
    if (lane == 0) {
        out[pix]        = a + head_b[idx * 2];
        out[SPIX + pix] = b + head_b[idx * 2 + 1];
    }
}

// =====================================================================
// Host launcher
// =====================================================================
static inline void launch_gemm(hipStream_t st,
                               const bf16* A, long long sA, int lda,
                               const bf16* B, long long sB, int ldb,
                               void* C, long long sC, int ldc,
                               int M, int N, int K, int mode, int batch)
{
    dim3 g((M + 63) / 64, (N + 63) / 64, batch);
    gemm_bf16_wmma<<<g, dim3(128), 0, st>>>(A, sA, lda, B, sB, ldb, C, sC, ldc, M, N, K, mode);
}

extern "C" void kernel_launch(void* const* d_in, const int* in_sizes, int n_in,
                              void* d_out, int out_size, void* d_ws, size_t ws_size,
                              hipStream_t stream)
{
    (void)in_sizes; (void)n_in; (void)out_size; (void)ws_size;

    const float* x_in   = (const float*)d_in[0];
    const int*   gridp  = (const int*)  d_in[1];
    const float* in_w   = (const float*)d_in[2];
    const float* in_b   = (const float*)d_in[3];
    const float* four_w = (const float*)d_in[4];
    const float* lin_w  = (const float*)d_in[5];
    const float* lin_b  = (const float*)d_in[6];
    const float* ln_g   = (const float*)d_in[7];
    const float* ln_b   = (const float*)d_in[8];
    const float* head_w = (const float*)d_in[9];
    const float* head_b = (const float*)d_in[10];
    float* out = (float*)d_out;

    char* w = (char*)d_ws;
    size_t off = 0;
    auto carve = [&](size_t bytes) -> void* {
        void* p = w + off;
        off = (off + bytes + 255) & ~(size_t)255;
        return p;
    };
    bf16*  Fw    = (bf16*) carve((size_t)128 * 360 * 2);
    bf16*  Fh    = (bf16*) carve((size_t)64 * 180 * 2);
    bf16*  GhI   = (bf16*) carve((size_t)360 * 64 * 2);
    bf16*  Gw    = (bf16*) carve((size_t)360 * 128 * 2);
    bf16*  linT  = (bf16*) carve((size_t)CC * CC * 2);
    float* xact  = (float*)carve((size_t)SC * 4);
    bf16*  xb    = (bf16*) carve((size_t)SC * 2);
    float* x1    = (float*)carve((size_t)SC * 4);
    bf16*  Yw    = (bf16*) carve((size_t)1440 * 8192 * 2);
    float* C2    = (float*)carve((size_t)8 * 64 * 8192 * 4);
    bf16*  Zpack = (bf16*) carve((size_t)2048 * 8 * 128 * 2);
    bf16*  Wpack = (bf16*) carve((size_t)2048 * 128 * 128 * 2);
    bf16*  Mout  = (bf16*) carve((size_t)2048 * 8 * 128 * 2);
    bf16*  Bmat  = (bf16*) carve((size_t)8 * 64 * 4096 * 2);
    bf16*  CoutH = (bf16*) carve((size_t)8 * 360 * 4096 * 2);

    build_basis<<<dim3(495), dim3(256), 0, stream>>>(Fw, Fh, GhI, Gw);

    input_proj<<<dim3((unsigned)((SC + 255) / 256)), dim3(256), 0, stream>>>(
        x_in, in_w, in_b, xact, xb);

    for (int d = 0; d < DEPTH; ++d) {
        pack_linw<<<dim3(16), dim3(256), 0, stream>>>(lin_w + (size_t)d * CC * CC, linT);
        pack_wmix<<<dim3(131072), dim3(256), 0, stream>>>(
            four_w + (size_t)d * CC * CC * ML * MW * 2, Wpack);

        // F1: W-forward DFT. per (b,h): [Fw(128x360)] @ [x slab (360x64)]
        launch_gemm(stream, Fw, 0, 360, xb, (long long)WW * CC, CC,
                    Yw, 8192, CC, 128, 64, 360, /*bf16*/1, BB * HH);

        // F2: H-forward DFT. per b: [Fh(64x180)] @ [Yw (180x8192)]
        launch_gemm(stream, Fh, 0, 180, Yw, (long long)HH * 8192, 8192,
                    C2, (long long)64 * 8192, 8192, 64, 8192, 180, /*f32*/0, BB);

        combine_Z<<<dim3(8192), dim3(256), 0, stream>>>(C2, Zpack);

        // MIX: per mode: [Z(8x128)] @ [Wpack(128x128)]
        launch_gemm(stream, Zpack, 1024, 128, Wpack, 16384, 128,
                    Mout, 1024, 128, 8, 128, 128, /*bf16*/1, 2048);

        pack_I1<<<dim3(8192), dim3(256), 0, stream>>>(Mout, Bmat);

        // I1: H-inverse DFT. per b: [GhI(360x64)] @ [Bmat(64x4096)]
        launch_gemm(stream, GhI, 0, 64, Bmat, (long long)64 * 4096, 4096,
                    CoutH, (long long)360 * 4096, 4096, 360, 4096, 64, /*bf16*/1, BB);

        // I2: W-inverse DFT. per (b,h): [Gw(360x128)] @ [slab(128x64)] -> x1 f32
        launch_gemm(stream, Gw, 0, 128, CoutH, 8192, CC,
                    x1, (long long)WW * CC, CC, 360, 64, 128, /*f32*/0, BB * HH);

        // LIN: x1 += xb @ linT  (518400x64x64), accumulate f32
        launch_gemm(stream, xb, 0, CC, linT, 0, CC,
                    x1, 0, CC, SPIX, CC, CC, /*acc*/2, 1);

        pw_block<<<dim3(SPIX / 8), dim3(256), 0, stream>>>(
            x1, lin_b + (size_t)d * CC, ln_g + (size_t)d * CC, ln_b + (size_t)d * CC,
            xact, xb);
    }

    head_block<<<dim3(SPIX / 8), dim3(256), 0, stream>>>(xact, gridp, head_w, head_b, out);
}
